// EnhancedTemporalSNN_DGCNN_fd_77223511982139
// MI455X (gfx1250) — matrix operand order repaired and checked by reading, not compiled, not run on hardware
//
#include <hip/hip_runtime.h>
#include <hip/hip_bf16.h>

typedef __attribute__((ext_vector_type(16))) _Float16 v16h;
typedef __attribute__((ext_vector_type(8)))  _Float16 v8h;
typedef __attribute__((ext_vector_type(8)))  float    v8f;
typedef unsigned int u32x4 __attribute__((ext_vector_type(4)));
typedef int          i32x4 __attribute__((ext_vector_type(4)));
typedef int          i32x8 __attribute__((ext_vector_type(8)));

static constexpr int Bsz = 2;
static constexpr int Npt = 1024;
static constexpr int Knb = 20;
static constexpr int Tst = 3;
static constexpr float EPSbn = 1e-5f;

#if __has_builtin(__builtin_amdgcn_tensor_load_to_lds) && \
    __has_builtin(__builtin_amdgcn_s_wait_tensorcnt)
#define USE_TDM 1
#else
#define USE_TDM 0
#endif

#if USE_TDM
// Issue a TDM 2D tile load: 64 rows x 32 f16 elems, row stride K elems,
// from global 'gsrc' into LDS byte offset 'ldsOff'. D# per CDNA5 ISA ch.8.
__device__ __forceinline__ void tdm_issue(unsigned ldsOff, const _Float16* gsrc, int K) {
  unsigned long long ga = (unsigned long long)(size_t)gsrc;
  u32x4 g0;
  g0[0] = 1u;                                             // count=1 (valid), user mode
  g0[1] = ldsOff;                                         // lds_addr (bytes)
  g0[2] = (unsigned)ga;                                   // global_addr[31:0]
  g0[3] = (unsigned)((ga >> 32) & 0x01FFFFFFu) | (2u << 30); // addr[56:32] | type=2
  i32x8 g1;
  g1[0] = (int)(1u << 16);                 // wg_mask=0, data_size=1 (2 bytes)
  g1[1] = (int)(32u << 16);                // tensor_dim0[15:0]=32 in bits[31:16]
  g1[2] = (int)(64u << 16);                // tensor_dim0 hi=0 | tensor_dim1 lo=64
  g1[3] = (int)(32u << 16);                // tensor_dim1 hi=0 | tile_dim0=32
  g1[4] = (int)64u;                        // tile_dim1=64 | tile_dim2=0
  g1[5] = (int)(unsigned)K;                // tensor_dim0_stride[31:0] = K elems
  g1[6] = 0;                               // stride0 hi | stride1 lo
  g1[7] = 0;                               // stride1 hi
  i32x4 z4 = {0, 0, 0, 0};
#if defined(__clang_major__) && __clang_major__ >= 23
  i32x8 z8 = {0, 0, 0, 0, 0, 0, 0, 0};
  __builtin_amdgcn_tensor_load_to_lds(g0, g1, z4, z4, z8, 0);
#else
  __builtin_amdgcn_tensor_load_to_lds(g0, g1, z4, z4, 0);
#endif
}
#else
// Fallback: cooperative synchronous tile copy (128 threads, 16B chunks)
__device__ __forceinline__ void copy_tile(_Float16* dst, const _Float16* src,
                                          int K, int tid) {
  for (int e = tid; e < 256; e += 128) {
    int row = e >> 2, part = (e & 3) << 3;
    *(v8h*)(dst + row * 32 + part) = *(const v8h*)(src + (size_t)row * K + part);
  }
}
#endif

// ---------------------------------------------------------------------------
// Workgroup WMMA GEMM: C[M,N] = A[M,K] * B[N,K]^T  (f16 in, f32 out)
// Block = 128 threads (4 waves) computing a 64x64 tile.
// A tile (64x32) staged in LDS via TDM (double buffered, overlapped with
// compute); B fragments loaded directly from global (L2-resident weights).
// Fragment layouts per CDNA5 ISA 7.12.2 (wave32):
//   A 16x32:  lane l: M=l&15; halfs [kk+(l>>4)*8 .. +7] and [+16 ..]
//   B 32x16:  lane l: N=l&15; halfs [kk+(l>>4)*8 .. +7] and [+16 ..] of row N of B
//   C 16x16:  vgpr r, lane l: M = r + (l>>4)*8, N = l&15
// Requires: K %32==0, M %64==0, N %64==0.
// ---------------------------------------------------------------------------
__global__ __launch_bounds__(128)
void wmma_gemm_tile(const _Float16* __restrict__ A, const _Float16* __restrict__ Bm,
                    float* __restrict__ C, int M, int N, int K) {
  const int tid  = threadIdx.x;
  const int wv   = tid >> 5;
  const int lane = tid & 31;
  const int l15  = lane & 15;
  const int hs   = lane >> 4;
  const int m0   = blockIdx.x << 6;   // M tiles on x (can be large)
  const int n0   = blockIdx.y << 6;

  __shared__ _Float16 shA[2][64 * 32];
  const unsigned TILE_BYTES = 64 * 32 * 2;
#if USE_TDM
  const unsigned ldsBase = (unsigned)(size_t)(const void*)&shA[0][0];
#endif
  const _Float16* Atile = A + (size_t)m0 * K;
  const _Float16* brow  = Bm + (size_t)(n0 + l15) * K + (hs << 3);
  (void)TILE_BYTES;

  // prologue: first A tile into buffer 0
#if USE_TDM
  if (tid < 32) {
    tdm_issue(ldsBase, Atile, K);
    __builtin_amdgcn_s_wait_tensorcnt(0);
  }
#else
  copy_tile(&shA[0][0], Atile, K, tid);
#endif
  __syncthreads();

  v8f acc[4] = {};
  for (int kk = 0; kk < K; kk += 32) {
    const int buf = (kk >> 5) & 1;
    const bool more = (kk + 32) < K;
    // start async fetch of next A tile into the other buffer
#if USE_TDM
    if (more && tid < 32)
      tdm_issue(ldsBase + (unsigned)(buf ^ 1) * TILE_BYTES, Atile + kk + 32, K);
#else
    if (more) copy_tile(&shA[buf ^ 1][0], Atile + kk + 32, K, tid);
#endif
    // A fragment from LDS (ds_load_b128 x2)
    const _Float16* arow = &shA[buf][(wv * 16 + l15) * 32 + (hs << 3)];
    v8h alo = *(const v8h*)arow;
    v8h ahi = *(const v8h*)(arow + 16);
    v16h af = __builtin_shufflevector(alo, ahi,
              0,1,2,3,4,5,6,7,8,9,10,11,12,13,14,15);
#pragma unroll
    for (int j = 0; j < 4; ++j) {
      const _Float16* bp = brow + (size_t)j * 16 * K + kk;
      if (more) __builtin_prefetch(bp + 32, 0, 3);
      v8h blo = *(const v8h*)bp;
      v8h bhi = *(const v8h*)(bp + 16);
      v16h bf = __builtin_shufflevector(blo, bhi,
                0,1,2,3,4,5,6,7,8,9,10,11,12,13,14,15);
      acc[j] = __builtin_amdgcn_wmma_f32_16x16x32_f16(
                 false, af, false, bf, (short)0, acc[j], false, false);
    }
#if USE_TDM
    if (more && tid < 32) __builtin_amdgcn_s_wait_tensorcnt(0);
#endif
    __syncthreads();
  }
#pragma unroll
  for (int j = 0; j < 4; ++j)
#pragma unroll
    for (int r = 0; r < 8; ++r)
      C[(size_t)(m0 + wv * 16 + (hs << 3) + r) * N + (n0 + j * 16 + l15)] = acc[j][r];
}

// f32 -> f16 convert with zero-padding of the K dimension
__global__ void cvt_pad_f16(const float* __restrict__ W, _Float16* __restrict__ Wh,
                            int O, int C, int Cpad) {
  size_t t = (size_t)blockIdx.x * blockDim.x + threadIdx.x;
  size_t tot = (size_t)O * Cpad;
  if (t >= tot) return;
  int o = (int)(t / Cpad), c = (int)(t % Cpad);
  Wh[t] = (c < C) ? (_Float16)W[(size_t)o * C + c] : (_Float16)0.f;
}

// x (B,3,N) -> feat0 (B,N,32) f16, pad zeros
__global__ void init_feat0(const float* __restrict__ x, _Float16* __restrict__ f) {
  size_t t = (size_t)blockIdx.x * blockDim.x + threadIdx.x;
  size_t tot = (size_t)Bsz * Npt * 32;
  if (t >= tot) return;
  int c = (int)(t & 31);
  size_t bn = t >> 5;
  int b = (int)(bn / Npt), n = (int)(bn % Npt);
  f[t] = (c < 3) ? (_Float16)x[((size_t)b * 3 + c) * Npt + n] : (_Float16)0.f;
}

// per-point squared norm over the (zero-padded) f16 feature row
__global__ void sumsq_kernel(const _Float16* __restrict__ feat, float* __restrict__ x2,
                             int C, int Cpad, int BN) {
  int t = blockIdx.x * blockDim.x + threadIdx.x;
  if (t >= BN) return;
  const _Float16* r = feat + (size_t)t * Cpad;
  float s = 0.f;
  for (int c = 0; c < C; ++c) { float v = (float)r[c]; s += v * v; }
  x2[t] = s;
}

// top-20 of neg_d = 2*inner - x2_i - x2_j  per (b, i); 128 threads/block
__global__ __launch_bounds__(128)
void topk_kernel(const float* __restrict__ inner, const float* __restrict__ x2,
                 int* __restrict__ idx) {
  const int i = blockIdx.x, b = blockIdx.y, tid = threadIdx.x;
  const float* row = inner + ((size_t)b * Npt + i) * Npt;
  const float xi = x2[b * Npt + i];
  float cand[8];
  unsigned picked = 0;
#pragma unroll
  for (int c = 0; c < 8; ++c) {
    int j = c * 128 + tid;
    cand[c] = 2.f * row[j] - xi - x2[b * Npt + j];
  }
  __shared__ float sv[128];
  __shared__ int   si[128];
  for (int sel = 0; sel < Knb; ++sel) {
    float bv = -3.4e38f; int bj = -1;
#pragma unroll
    for (int c = 0; c < 8; ++c)
      if (!((picked >> c) & 1u) && cand[c] > bv) { bv = cand[c]; bj = c * 128 + tid; }
    sv[tid] = bv; si[tid] = bj;
    __syncthreads();
    for (int s = 64; s > 0; s >>= 1) {
      if (tid < s && sv[tid + s] > sv[tid]) { sv[tid] = sv[tid + s]; si[tid] = si[tid + s]; }
      __syncthreads();
    }
    int win = si[0];
    if (tid == 0) idx[((size_t)b * Npt + i) * Knb + sel] = win;
    if ((win & 127) == tid) picked |= 1u << (win >> 7);
    __syncthreads();
  }
}

// graph feature rows: gf[(b,n,k), :] = [nb - ctr (C), nb (C), 0 pad] as f16
__global__ void build_gf(const _Float16* __restrict__ feat, const int* __restrict__ idx,
                         _Float16* __restrict__ gf, int C, int CpadF, int Kpad) {
  size_t t = (size_t)blockIdx.x * blockDim.x + threadIdx.x;
  size_t tot = (size_t)Bsz * Npt * Knb * Kpad;
  if (t >= tot) return;
  size_t row = t / Kpad;
  int c = (int)(t % Kpad);
  int b = (int)(row / ((size_t)Npt * Knb));
  int n = (int)((row / Knb) % Npt);
  int j = idx[row];
  _Float16 v = (_Float16)0.f;
  if (c < C) {
    v = (_Float16)((float)feat[((size_t)b * Npt + j) * CpadF + c] -
                   (float)feat[((size_t)b * Npt + n) * CpadF + c]);
  } else if (c < 2 * C) {
    v = feat[((size_t)b * Npt + j) * CpadF + (c - C)];
  }
  gf[t] = v;
}

// column mean/var of H[M, Nc]; one block per channel
__global__ __launch_bounds__(256)
void colstats(const float* __restrict__ H, int M, int Nc,
              float* __restrict__ mean, float* __restrict__ var) {
  int o = blockIdx.x, tid = threadIdx.x;
  float s = 0.f, s2 = 0.f;
  for (int r = tid; r < M; r += 256) {
    float v = H[(size_t)r * Nc + o];
    s += v; s2 += v * v;
  }
  __shared__ float ss[256], sq[256];
  ss[tid] = s; sq[tid] = s2;
  __syncthreads();
  for (int st = 128; st > 0; st >>= 1) {
    if (tid < st) { ss[tid] += ss[tid + st]; sq[tid] += sq[tid + st]; }
    __syncthreads();
  }
  if (tid == 0) {
    float m = ss[0] / (float)M;
    mean[o] = m;
    var[o]  = sq[0] / (float)M - m * m;
  }
}

__device__ __forceinline__ float clampf(float x, float lo, float hi) {
  return fminf(fmaxf(x, lo), hi);
}

// BN + lrelu + max over k + LIF update; writes binary spike to feat/cat (f16)
__global__ void bn_max_lif(const float* __restrict__ H,
                           const float* __restrict__ mean, const float* __restrict__ var,
                           const float* __restrict__ g, const float* __restrict__ bta,
                           int Cout, int first,
                           const float* __restrict__ md, const float* __restrict__ ta,
                           const float* __restrict__ rd, const float* __restrict__ tb,
                           float* __restrict__ mem, float* __restrict__ thr,
                           float* __restrict__ ref,
                           _Float16* __restrict__ feat_out, int CpadOut,
                           _Float16* __restrict__ cat_out, int catOff) {
  size_t t = (size_t)blockIdx.x * blockDim.x + threadIdx.x;
  size_t tot = (size_t)Bsz * Npt * Cout;
  if (t >= tot) return;
  int o = (int)(t % Cout);
  size_t bn = t / Cout;
  float mn = mean[o], iv = rsqrtf(var[o] + EPSbn), gg = g[o], bo = bta[o];
  const float* hrow = H + bn * (size_t)Knb * Cout + o;
  float mx = -3.4e38f;
  for (int k = 0; k < Knb; ++k) {
    float v = (hrow[(size_t)k * Cout] - mn) * iv * gg + bo;
    v = v >= 0.f ? v : 0.2f * v;
    mx = fmaxf(mx, v);
  }
  float m_d = clampf(md[o], 0.1f, 0.99f);
  float t_a = clampf(ta[o], 0.001f, 0.1f);
  float r_d = clampf(rd[o], 0.1f, 0.95f);
  float t_b = tb[o];
  float mm, tt, rr;
  if (first) { mm = 0.f; tt = t_b; rr = 0.f; }
  else       { mm = mem[t]; tt = thr[t]; rr = ref[t]; }
  float xin = mx * (rr <= 0.f ? 1.f : 0.f);
  mm = mm * m_d * (1.f - rr) + xin;
  float s = (mm - tt) > 0.f ? 1.f : 0.f;   // forward value of surrogate spike
  mm = mm * (1.f - s);
  rr = rr * r_d + s;
  tt = t_b + (tt + t_a * s - t_b) * 0.95f;
  mem[t] = mm; thr[t] = tt; ref[t] = rr;
  if (feat_out) feat_out[bn * CpadOut + o] = (_Float16)s;
  cat_out[bn * 960 + catOff + o] = (_Float16)s;
}

// BN + lrelu + global max over N; one block per (channel, batch)
__global__ __launch_bounds__(256)
void agg_bn_pool(const float* __restrict__ agg,
                 const float* __restrict__ mean, const float* __restrict__ var,
                 const float* __restrict__ g, const float* __restrict__ b,
                 float* __restrict__ pooled, int Cout) {
  int o = blockIdx.x, bb = blockIdx.y, tid = threadIdx.x;
  float mn = mean[o], iv = rsqrtf(var[o] + EPSbn), gg = g[o], bo = b[o];
  float mx = -3.4e38f;
  for (int n = tid; n < Npt; n += 256) {
    float v = (agg[((size_t)bb * Npt + n) * Cout + o] - mn) * iv * gg + bo;
    v = v >= 0.f ? v : 0.2f * v;
    mx = fmaxf(mx, v);
  }
  __shared__ float sm[256];
  sm[tid] = mx; __syncthreads();
  for (int s = 128; s > 0; s >>= 1) {
    if (tid < s) sm[tid] = fmaxf(sm[tid], sm[tid + s]);
    __syncthreads();
  }
  if (tid == 0) pooled[(size_t)bb * Cout + o] = sm[0];
}

// temporal softmax mix + final fresh-state LIF (forward value = hard spike)
__global__ void final_kernel(const float* __restrict__ pooled, const float* __restrict__ tw,
                             const float* __restrict__ lftb, float* __restrict__ out) {
  int t = blockIdx.x * blockDim.x + threadIdx.x;
  const int F = 512;
  if (t >= Bsz * F) return;
  int f = t % F;
  float w0 = tw[0], w1 = tw[1], w2 = tw[2];
  float m = fmaxf(w0, fmaxf(w1, w2));
  float e0 = __expf(w0 - m), e1 = __expf(w1 - m), e2 = __expf(w2 - m);
  float inv = 1.f / (e0 + e1 + e2);
  float o = (e0 * pooled[t] + e1 * pooled[Bsz * F + t] + e2 * pooled[2 * Bsz * F + t]) * inv;
  out[t] = (o - lftb[f]) > 0.f ? 1.f : 0.f;
}

// ---------------------------------------------------------------------------
extern "C" void kernel_launch(void* const* d_in, const int* in_sizes, int n_in,
                              void* d_out, int out_size, void* d_ws, size_t ws_size,
                              hipStream_t stream) {
  (void)in_sizes; (void)n_in; (void)out_size; (void)ws_size;
  const float* x  = (const float*)d_in[0];
  const float* W[4]  = {(const float*)d_in[1], (const float*)d_in[4],
                        (const float*)d_in[7], (const float*)d_in[10]};
  const float* gP[4] = {(const float*)d_in[2], (const float*)d_in[5],
                        (const float*)d_in[8], (const float*)d_in[11]};
  const float* bP[4] = {(const float*)d_in[3], (const float*)d_in[6],
                        (const float*)d_in[9], (const float*)d_in[12]};
  const float* Wm = (const float*)d_in[13];
  const float* gm = (const float*)d_in[14];
  const float* bm = (const float*)d_in[15];
  const float* md[4], *ta[4], *rd[4], *tb[4];
  for (int i = 0; i < 4; ++i) {
    md[i] = (const float*)d_in[16 + 4 * i + 0];
    ta[i] = (const float*)d_in[16 + 4 * i + 1];
    rd[i] = (const float*)d_in[16 + 4 * i + 2];
    tb[i] = (const float*)d_in[16 + 4 * i + 3];
  }
  const float* lftb = (const float*)d_in[35];
  const float* tw   = (const float*)d_in[36];
  float* out = (float*)d_out;

  const int Cin[4]    = {3, 64, 128, 256};
  const int CpadIn[4] = {32, 64, 128, 256};
  const int Kgf[4]    = {32, 128, 256, 512};
  const int Cout[4]   = {64, 128, 256, 512};
  const int catOff[4] = {0, 64, 192, 448};
  const int BN  = Bsz * Npt;       // 2048
  const int Mgf = BN * Knb;        // 40960

  // --- workspace carve (256B aligned) ---
  char* p = (char*)d_ws;
  auto alloc = [&](size_t bytes) -> void* {
    void* r = (void*)p;
    p += (bytes + 255) & ~(size_t)255;
    return r;
  };
  _Float16* Wh[4];
  Wh[0] = (_Float16*)alloc((size_t)64 * 32 * 2);
  Wh[1] = (_Float16*)alloc((size_t)128 * 128 * 2);
  Wh[2] = (_Float16*)alloc((size_t)256 * 256 * 2);
  Wh[3] = (_Float16*)alloc((size_t)512 * 512 * 2);
  _Float16* Wmh   = (_Float16*)alloc((size_t)512 * 960 * 2);
  _Float16* feat0 = (_Float16*)alloc((size_t)BN * 32 * 2);
  _Float16* featO[4] = {nullptr, nullptr, nullptr, nullptr};
  for (int i = 0; i < 3; ++i) featO[i] = (_Float16*)alloc((size_t)BN * Cout[i] * 2);
  _Float16* catB  = (_Float16*)alloc((size_t)BN * 960 * 2);
  float* x2    = (float*)alloc((size_t)BN * 4);
  float* inner = (float*)alloc((size_t)Bsz * Npt * Npt * 4);
  int*   idx   = (int*)alloc((size_t)BN * Knb * 4);
  _Float16* gf = (_Float16*)alloc((size_t)Mgf * 512 * 2);
  float* Hbuf  = (float*)alloc((size_t)Mgf * 512 * 4);
  float* meanB = (float*)alloc(512 * 4);
  float* varB  = (float*)alloc(512 * 4);
  float* memS[4], *thrS[4], *refS[4];
  for (int i = 0; i < 4; ++i) {
    memS[i] = (float*)alloc((size_t)BN * Cout[i] * 4);
    thrS[i] = (float*)alloc((size_t)BN * Cout[i] * 4);
    refS[i] = (float*)alloc((size_t)BN * Cout[i] * 4);
  }
  float* agg    = (float*)alloc((size_t)BN * 512 * 4);
  float* pooled = (float*)alloc((size_t)Tst * Bsz * 512 * 4);

  auto cdiv = [](size_t a, size_t b) { return (unsigned)((a + b - 1) / b); };

  // --- weight conversion (f32 -> padded f16) ---
  cvt_pad_f16<<<cdiv((size_t)64 * 32, 256), 256, 0, stream>>>(W[0], Wh[0], 64, 6, 32);
  cvt_pad_f16<<<cdiv((size_t)128 * 128, 256), 256, 0, stream>>>(W[1], Wh[1], 128, 128, 128);
  cvt_pad_f16<<<cdiv((size_t)256 * 256, 256), 256, 0, stream>>>(W[2], Wh[2], 256, 256, 256);
  cvt_pad_f16<<<cdiv((size_t)512 * 512, 256), 256, 0, stream>>>(W[3], Wh[3], 512, 512, 512);
  cvt_pad_f16<<<cdiv((size_t)512 * 960, 256), 256, 0, stream>>>(Wm, Wmh, 512, 960, 960);
  init_feat0<<<cdiv((size_t)BN * 32, 256), 256, 0, stream>>>(x, feat0);

  for (int t = 0; t < Tst; ++t) {
    const _Float16* cur = feat0;
    for (int i = 0; i < 4; ++i) {
      const int C = Cin[i], Cp = CpadIn[i], Kp = Kgf[i], Co = Cout[i];
      // kNN: squared norms + inner products (WMMA+TDM) + top-20
      sumsq_kernel<<<cdiv(BN, 256), 256, 0, stream>>>(cur, x2, C, Cp, BN);
      for (int b = 0; b < Bsz; ++b)
        wmma_gemm_tile<<<dim3(Npt / 64, Npt / 64), 128, 0, stream>>>(
            cur + (size_t)b * Npt * Cp, cur + (size_t)b * Npt * Cp,
            inner + (size_t)b * Npt * Npt, Npt, Npt, Cp);
      topk_kernel<<<dim3(Npt, Bsz), 128, 0, stream>>>(inner, x2, idx);
      // graph feature + edge-conv GEMM (WMMA+TDM)
      build_gf<<<cdiv((size_t)Mgf * Kp, 256), 256, 0, stream>>>(cur, idx, gf, C, Cp, Kp);
      wmma_gemm_tile<<<dim3(Mgf / 64, Co / 64), 128, 0, stream>>>(gf, Wh[i], Hbuf, Mgf, Co, Kp);
      // BN stats + fused BN/lrelu/max-k/LIF
      colstats<<<Co, 256, 0, stream>>>(Hbuf, Mgf, Co, meanB, varB);
      bn_max_lif<<<cdiv((size_t)BN * Co, 256), 256, 0, stream>>>(
          Hbuf, meanB, varB, gP[i], bP[i], Co, (t == 0) ? 1 : 0,
          md[i], ta[i], rd[i], tb[i], memS[i], thrS[i], refS[i],
          featO[i], Co, catB, catOff[i]);
      cur = featO[i];
    }
    // aggregation GEMM (WMMA+TDM) + BN + lrelu + global max-pool
    wmma_gemm_tile<<<dim3(BN / 64, 512 / 64), 128, 0, stream>>>(catB, Wmh, agg, BN, 512, 960);
    colstats<<<512, 256, 0, stream>>>(agg, BN, 512, meanB, varB);
    agg_bn_pool<<<dim3(512, Bsz), 256, 0, stream>>>(agg, meanB, varB, gm, bm,
                                                    pooled + (size_t)t * Bsz * 512, 512);
  }
  final_kernel<<<cdiv(Bsz * 512, 256), 256, 0, stream>>>(pooled, tw, lftb, out);
}
